// NeuroMUSX_V2_27401891349155
// MI455X (gfx1250) — matrix operand order repaired and checked by reading, not compiled, not run on hardware
//
#include <hip/hip_runtime.h>
#include <hip/hip_bf16.h>
#include <hip/hip_fp16.h>

// ---------------------------------------------------------------------------
// GATv2 GNN forward for MI455X (gfx1250, wave32).
//  - Node GEMMs (h@Wl, h@Wr, 20000x128x128) run on v_wmma_f32_16x16x32_f16.
//  - Edge softmax/aggregation (the memory-bound part) uses native f32/u32
//    global atomics with the monotone-uint float-max trick.
//  - Input pointer order: depth-first insertion order of setup_inputs().
// ---------------------------------------------------------------------------

typedef __attribute__((ext_vector_type(16))) _Float16 v16h;
typedef __attribute__((ext_vector_type(8)))  _Float16 v8h;
typedef __attribute__((ext_vector_type(8)))  float    v8f;

#define NODES   20000
#define EDGES   320000
#define NBATCH  16
#define DHID    128
#define NHEADS  16
#define NITER   10
#define NEGS    0.2f

__device__ __forceinline__ unsigned nmx_fkey(float f) {
  unsigned b = __float_as_uint(f);
  return (b & 0x80000000u) ? ~b : (b | 0x80000000u);
}
__device__ __forceinline__ float nmx_fdec(unsigned k) {
  return __uint_as_float((k & 0x80000000u) ? (k & 0x7FFFFFFFu) : ~k);
}

// ---------------- self-loop edge_attr (fill_value='mean') -------------------
__global__ __launch_bounds__(256) void nmx_deg(const int* __restrict__ dst,
                                               const float* __restrict__ ea,
                                               float* __restrict__ deg,
                                               float* __restrict__ ealoop, int E) {
  int e = blockIdx.x * blockDim.x + threadIdx.x;
  if (e >= E) return;
  int d = dst[e];
  atomicAdd(&deg[d], 1.0f);
  atomicAdd(&ealoop[2 * d + 0], ea[2 * e + 0]);
  atomicAdd(&ealoop[2 * d + 1], ea[2 * e + 1]);
}

__global__ __launch_bounds__(256) void nmx_ealoop_div(const float* __restrict__ deg,
                                                      float* __restrict__ ealoop, int n) {
  int i = blockIdx.x * blockDim.x + threadIdx.x;
  if (i >= n) return;
  float dv = fmaxf(deg[i], 1.0f);
  ealoop[2 * i + 0] /= dv;
  ealoop[2 * i + 1] /= dv;
}

// ---------------- init layer xl/xr (K=2, not worth WMMA) --------------------
__global__ __launch_bounds__(256) void nmx_init_xlxr(const float* __restrict__ x,
                                                     const float* __restrict__ Wl,
                                                     const float* __restrict__ bl,
                                                     const float* __restrict__ Wr,
                                                     const float* __restrict__ br,
                                                     float* __restrict__ xl,
                                                     float* __restrict__ xr, int n128) {
  int i = blockIdx.x * blockDim.x + threadIdx.x;
  if (i >= n128) return;
  int node = i >> 7, c = i & 127;
  float x0 = x[2 * node], x1 = x[2 * node + 1];
  xl[i] = x0 * Wl[c] + x1 * Wl[DHID + c] + bl[c];
  xr[i] = x0 * Wr[c] + x1 * Wr[DHID + c] + br[c];
}

// ---------------- fp16 conversion / weight transpose ------------------------
__global__ __launch_bounds__(256) void nmx_f32_to_f16(const float* __restrict__ in,
                                                      _Float16* __restrict__ out, int n) {
  int i = blockIdx.x * blockDim.x + threadIdx.x;
  if (i < n) out[i] = (_Float16)in[i];
}

// W: [128 (k), 128 (n)] row-major f32  ->  Wt: [n][k] f16
__global__ __launch_bounds__(256) void nmx_wtrans(const float* __restrict__ W,
                                                  _Float16* __restrict__ Wt) {
  int i = blockIdx.x * blockDim.x + threadIdx.x;
  if (i >= DHID * DHID) return;
  int k = i >> 7, n = i & 127;
  Wt[n * DHID + k] = (_Float16)W[i];
}

// ---------------- WMMA GEMM:  Out[N,128] = A_f16[N,128] @ W + bias ----------
// One wave per 16-row tile; 8 column tiles x 4 K-steps of v_wmma 16x16x32 f16.
__global__ __launch_bounds__(256) void nmx_gemm_wmma(const _Float16* __restrict__ A,
                                                     const _Float16* __restrict__ Wt,
                                                     const float* __restrict__ bias,
                                                     float* __restrict__ Out, int nrows) {
  int wid  = (blockIdx.x * blockDim.x + threadIdx.x) >> 5;   // row tile
  int lane = threadIdx.x & 31;
  int row0 = wid * 16;
  if (row0 >= nrows) return;                                  // wave-uniform exit
  int mrow = lane & 15;
  int half = lane >> 4;

  const _Float16* arow = A + (size_t)(row0 + mrow) * DHID;
  v8f acc[8] = {};
  union Frag { v16h v; v8h h[2]; };

#pragma unroll
  for (int kt = 0; kt < 4; ++kt) {
    int ka = kt * 32 + half * 8;          // A: elems 0..7 -> ka..ka+7 ; 8..15 -> ka+16..+23
    Frag a;
    a.h[0] = *(const v8h*)(arow + ka);
    a.h[1] = *(const v8h*)(arow + ka + 16);
    int kb = kt * 32 + half * 8;          // B: elems 0..7 -> kb..kb+7 ; 8..15 -> kb+16..+23
#pragma unroll
    for (int nt = 0; nt < 8; ++nt) {
      const _Float16* wrow = Wt + (size_t)(nt * 16 + mrow) * DHID + kb;
      Frag b;
      b.h[0] = *(const v8h*)(wrow);
      b.h[1] = *(const v8h*)(wrow + 16);
      acc[nt] = __builtin_amdgcn_wmma_f32_16x16x32_f16(
          false, a.v, false, b.v, (short)0, acc[nt], false, false);
    }
  }
#pragma unroll
  for (int nt = 0; nt < 8; ++nt) {
    float bb = bias[nt * 16 + mrow];
#pragma unroll
    for (int r = 0; r < 8; ++r) {
      Out[(size_t)(row0 + half * 8 + r) * DHID + nt * 16 + mrow] = acc[nt][r] + bb;
    }
  }
}

// ---------------- edge pass 1: logits + segment max (16 heads) --------------
__global__ __launch_bounds__(256) void nmx_edge_logit16(
    const int* __restrict__ src, const int* __restrict__ dst, int E0, int Eaug,
    const float* __restrict__ ea, const float* __restrict__ ealoop,
    const float* __restrict__ xl, const float* __restrict__ xr,
    const float* __restrict__ We, const float* __restrict__ att,
    float* __restrict__ logit, unsigned* __restrict__ mxkey) {
  __shared__ float sWe[2 * DHID];
  __shared__ float sAtt[DHID];
  for (int i = threadIdx.x; i < 2 * DHID; i += blockDim.x) sWe[i] = We[i];
  for (int i = threadIdx.x; i < DHID; i += blockDim.x) sAtt[i] = att[i];
  __syncthreads();

  int wid  = (blockIdx.x * blockDim.x + threadIdx.x) >> 5;   // one wave per edge
  int lane = threadIdx.x & 31;
  if (wid >= Eaug) return;

  int s, d; float e0, e1;
  if (wid < E0) { s = src[wid]; d = dst[wid]; e0 = ea[2 * wid]; e1 = ea[2 * wid + 1]; }
  else          { s = d = wid - E0; e0 = ealoop[2 * s]; e1 = ealoop[2 * s + 1]; }

  int c0 = lane * 4;                       // 4 channels/lane, head = lane>>1
  float4 vl = *(const float4*)(xl + (size_t)s * DHID + c0);
  float4 vr = *(const float4*)(xr + (size_t)d * DHID + c0);
  float lv[4] = {vl.x + vr.x, vl.y + vr.y, vl.z + vr.z, vl.w + vr.w};
  float acc = 0.f;
#pragma unroll
  for (int k = 0; k < 4; ++k) {
    int c = c0 + k;
    float v = lv[k] + e0 * sWe[c] + e1 * sWe[DHID + c];
    v = v > 0.f ? v : NEGS * v;
    acc += v * sAtt[c];
  }
  acc += __shfl_xor(acc, 1, 32);           // pair-reduce -> per-head logit
  if ((lane & 1) == 0) {
    int h = lane >> 1;
    logit[(size_t)wid * NHEADS + h] = acc;
    atomicMax(&mxkey[(size_t)d * NHEADS + h], nmx_fkey(acc));
  }
}

// ---------------- edge pass 2: exp + denom ---------------------------------
__global__ __launch_bounds__(256) void nmx_edge_soft16(
    const int* __restrict__ dst, int E0, int Eaug,
    float* __restrict__ logit, const unsigned* __restrict__ mxkey,
    float* __restrict__ den) {
  int t = blockIdx.x * blockDim.x + threadIdx.x;    // t = e*16 + h
  if (t >= Eaug * NHEADS) return;
  int e = t >> 4, h = t & 15;
  int d = (e < E0) ? dst[e] : (e - E0);
  float ex = expf(logit[t] - nmx_fdec(mxkey[(size_t)d * NHEADS + h]));
  logit[t] = ex;
  atomicAdd(&den[(size_t)d * NHEADS + h], ex);
}

// ---------------- edge pass 3: alpha-weighted scatter of xl[src] ------------
__global__ __launch_bounds__(256) void nmx_edge_agg16(
    const int* __restrict__ src, const int* __restrict__ dst, int E0, int Eaug,
    const float* __restrict__ ex, const float* __restrict__ den,
    const float* __restrict__ xl, float* __restrict__ out) {
  int wid  = (blockIdx.x * blockDim.x + threadIdx.x) >> 5;
  int lane = threadIdx.x & 31;
  if (wid >= Eaug) return;
  int s, d;
  if (wid < E0) { s = src[wid]; d = dst[wid]; } else { s = d = wid - E0; }
  int h = lane >> 1;
  float alpha = ex[(size_t)wid * NHEADS + h] /
                (den[(size_t)d * NHEADS + h] + 1e-16f);
  int c0 = lane * 4;
  float4 v = *(const float4*)(xl + (size_t)s * DHID + c0);
  float* od = out + (size_t)d * DHID + c0;
  atomicAdd(od + 0, alpha * v.x);
  atomicAdd(od + 1, alpha * v.y);
  atomicAdd(od + 2, alpha * v.z);
  atomicAdd(od + 3, alpha * v.w);
}

// ---------------- bias + BN + ELU (+ residual) ------------------------------
__global__ __launch_bounds__(256) void nmx_bn_elu(
    const float* __restrict__ agg, const float* __restrict__ bias,
    const float* __restrict__ gamma, const float* __restrict__ beta,
    const float* __restrict__ mean, const float* __restrict__ var,
    const float* __restrict__ h0, int addres,
    float* __restrict__ hout, float* __restrict__ h0out, int n128) {
  int i = blockIdx.x * blockDim.x + threadIdx.x;
  if (i >= n128) return;
  int c = i & 127;
  float y = agg[i] + bias[c];
  y = (y - mean[c]) * rsqrtf(var[c] + 1e-5f) * gamma[c] + beta[c];
  y = y > 0.f ? y : (expf(y) - 1.f);           // ELU(alpha=1)
  if (addres) y += h0[i];
  hout[i] = y;
  if (h0out) h0out[i] = y;
}

// ---------------- final 1-head GEMVs (mus & sat, 4 dots/node) ---------------
__global__ __launch_bounds__(256) void nmx_final_gemv(
    const float* __restrict__ h,
    const float* __restrict__ WlM, const float* __restrict__ WrM,
    const float* __restrict__ WlS, const float* __restrict__ WrS,
    const float* __restrict__ blM, const float* __restrict__ brM,
    const float* __restrict__ blS, const float* __restrict__ brS,
    float* __restrict__ xlm, float* __restrict__ xrm,
    float* __restrict__ xls, float* __restrict__ xrs, int n) {
  int wid  = (blockIdx.x * blockDim.x + threadIdx.x) >> 5;   // one wave per node
  int lane = threadIdx.x & 31;
  if (wid >= n) return;
  const float* hr = h + (size_t)wid * DHID;
  float a = 0.f, b = 0.f, c = 0.f, d = 0.f;
  for (int k = lane; k < DHID; k += 32) {
    float hv = hr[k];
    a += hv * WlM[k]; b += hv * WrM[k];
    c += hv * WlS[k]; d += hv * WrS[k];
  }
#pragma unroll
  for (int off = 16; off > 0; off >>= 1) {
    a += __shfl_xor(a, off, 32); b += __shfl_xor(b, off, 32);
    c += __shfl_xor(c, off, 32); d += __shfl_xor(d, off, 32);
  }
  if (lane == 0) {
    xlm[wid] = a + blM[0]; xrm[wid] = b + brM[0];
    xls[wid] = c + blS[0]; xrs[wid] = d + brS[0];
  }
}

// ---------------- final edge passes (1 head, mus & sat together) ------------
__global__ __launch_bounds__(256) void nmx_final_logit(
    const int* __restrict__ src, const int* __restrict__ dst, int E0, int Eaug,
    const float* __restrict__ ea, const float* __restrict__ ealoop,
    const float* __restrict__ xlm, const float* __restrict__ xrm,
    const float* __restrict__ Wem, const float* __restrict__ attm,
    const float* __restrict__ xls, const float* __restrict__ xrs,
    const float* __restrict__ Wes, const float* __restrict__ atts,
    float* __restrict__ lm, float* __restrict__ ls,
    unsigned* __restrict__ mxm, unsigned* __restrict__ mxs) {
  int e = blockIdx.x * blockDim.x + threadIdx.x;
  if (e >= Eaug) return;
  int s, d; float e0, e1;
  if (e < E0) { s = src[e]; d = dst[e]; e0 = ea[2 * e]; e1 = ea[2 * e + 1]; }
  else        { s = d = e - E0; e0 = ealoop[2 * s]; e1 = ealoop[2 * s + 1]; }
  float vm = xlm[s] + xrm[d] + e0 * Wem[0] + e1 * Wem[1];
  vm = vm > 0.f ? vm : NEGS * vm;
  float Lm = vm * attm[0];
  float vs = xls[s] + xrs[d] + e0 * Wes[0] + e1 * Wes[1];
  vs = vs > 0.f ? vs : NEGS * vs;
  float Ls = vs * atts[0];
  lm[e] = Lm; ls[e] = Ls;
  atomicMax(&mxm[d], nmx_fkey(Lm));
  atomicMax(&mxs[d], nmx_fkey(Ls));
}

__global__ __launch_bounds__(256) void nmx_final_soft(
    const int* __restrict__ dst, int E0, int Eaug,
    float* __restrict__ lm, float* __restrict__ ls,
    const unsigned* __restrict__ mxm, const unsigned* __restrict__ mxs,
    float* __restrict__ denm, float* __restrict__ dens) {
  int e = blockIdx.x * blockDim.x + threadIdx.x;
  if (e >= Eaug) return;
  int d = (e < E0) ? dst[e] : (e - E0);
  float em = expf(lm[e] - nmx_fdec(mxm[d])); lm[e] = em; atomicAdd(&denm[d], em);
  float es = expf(ls[e] - nmx_fdec(mxs[d])); ls[e] = es; atomicAdd(&dens[d], es);
}

__global__ __launch_bounds__(256) void nmx_final_agg(
    const int* __restrict__ src, const int* __restrict__ dst, int E0, int Eaug,
    const float* __restrict__ lm, const float* __restrict__ ls,
    const float* __restrict__ denm, const float* __restrict__ dens,
    const float* __restrict__ xlm, const float* __restrict__ xls,
    float* __restrict__ outm, float* __restrict__ outs) {
  int e = blockIdx.x * blockDim.x + threadIdx.x;
  if (e >= Eaug) return;
  int s, d;
  if (e < E0) { s = src[e]; d = dst[e]; } else { s = d = e - E0; }
  atomicAdd(&outm[d], lm[e] / (denm[d] + 1e-16f) * xlm[s]);
  atomicAdd(&outs[d], ls[e] / (dens[d] + 1e-16f) * xls[s]);
}

// ---------------- outputs ---------------------------------------------------
__global__ __launch_bounds__(256) void nmx_out_nodes(
    const float* __restrict__ outm, const float* __restrict__ outs,
    const float* __restrict__ biasM, const float* __restrict__ biasS,
    const int* __restrict__ mask, const int* __restrict__ batch,
    float* __restrict__ dout, float* __restrict__ sums,
    float* __restrict__ cnt, int n) {
  int i = blockIdx.x * blockDim.x + threadIdx.x;
  if (i >= n) return;
  dout[i] = outm[i] + biasM[0];
  float satv = outs[i] + biasS[0];
  if (mask[i] == 0) {
    atomicAdd(&sums[batch[i]], satv);
    atomicAdd(&cnt[batch[i]], 1.0f);
  }
}

__global__ void nmx_out_batch(const float* __restrict__ sums,
                              const float* __restrict__ cnt,
                              float* __restrict__ dout, int n, int nb) {
  int b = blockIdx.x * blockDim.x + threadIdx.x;
  if (b >= nb) return;
  dout[n + b] = sums[b] / fmaxf(cnt[b], 1.0f);
}

// ===========================================================================
extern "C" void kernel_launch(void* const* d_in, const int* in_sizes, int n_in,
                              void* d_out, int out_size, void* d_ws, size_t ws_size,
                              hipStream_t stream) {
  (void)in_sizes; (void)n_in; (void)out_size; (void)ws_size;
  const int N = NODES, E = EDGES, Eaug = EDGES + NODES;
  auto cdiv = [](long long a, long long b) { return (int)((a + b - 1) / b); };

  // ---- inputs (depth-first insertion order of setup_inputs dict) ----
  const float* x     = (const float*)d_in[0];
  const int*   ei    = (const int*)d_in[1];
  const int*   src   = ei;
  const int*   dst   = ei + E;
  const float* eattr = (const float*)d_in[2];
  const int*   mask  = (const int*)d_in[3];
  const int*   batch = (const int*)d_in[4];
  int p = 5;
  // init_gat
  const float* iWl = (const float*)d_in[p++]; const float* ibl = (const float*)d_in[p++];
  const float* iWr = (const float*)d_in[p++]; const float* ibr = (const float*)d_in[p++];
  const float* iWe = (const float*)d_in[p++]; const float* iatt = (const float*)d_in[p++];
  const float* ibias = (const float*)d_in[p++];
  // init_bn
  const float* ig = (const float*)d_in[p++]; const float* ibe = (const float*)d_in[p++];
  const float* imn = (const float*)d_in[p++]; const float* ivr = (const float*)d_in[p++];
  // hid_gat (leading dim NITER)
  const float* hWl = (const float*)d_in[p++]; const float* hbl = (const float*)d_in[p++];
  const float* hWr = (const float*)d_in[p++]; const float* hbr = (const float*)d_in[p++];
  const float* hWe = (const float*)d_in[p++]; const float* hatt = (const float*)d_in[p++];
  const float* hbias = (const float*)d_in[p++];
  // hid_bn
  const float* hg = (const float*)d_in[p++]; const float* hbe = (const float*)d_in[p++];
  const float* hmn = (const float*)d_in[p++]; const float* hvr = (const float*)d_in[p++];
  // mus
  const float* mWl = (const float*)d_in[p++]; const float* mbl = (const float*)d_in[p++];
  const float* mWr = (const float*)d_in[p++]; const float* mbr = (const float*)d_in[p++];
  const float* mWe = (const float*)d_in[p++]; const float* matt = (const float*)d_in[p++];
  const float* mbias = (const float*)d_in[p++];
  // sat
  const float* sWl = (const float*)d_in[p++]; const float* sbl = (const float*)d_in[p++];
  const float* sWr = (const float*)d_in[p++]; const float* sbr = (const float*)d_in[p++];
  const float* sWe = (const float*)d_in[p++]; const float* satt = (const float*)d_in[p++];
  const float* sbias = (const float*)d_in[p++];

  // ---- workspace carve ----
  char* wp = (char*)d_ws;
  auto carve = [&](size_t bytes) -> void* {
    void* r = (void*)wp;
    wp += (bytes + 255) & ~(size_t)255;
    return r;
  };
  float*     deg    = (float*)carve((size_t)N * 4);
  float*     ealoop = (float*)carve((size_t)N * 2 * 4);
  float*     h      = (float*)carve((size_t)N * DHID * 4);
  float*     h0     = (float*)carve((size_t)N * DHID * 4);
  _Float16*  hf16   = (_Float16*)carve((size_t)N * DHID * 2);
  _Float16*  WlT    = (_Float16*)carve((size_t)DHID * DHID * 2);
  _Float16*  WrT    = (_Float16*)carve((size_t)DHID * DHID * 2);
  float*     xl     = (float*)carve((size_t)N * DHID * 4);
  float*     xr     = (float*)carve((size_t)N * DHID * 4);
  float*     agg    = (float*)carve((size_t)N * DHID * 4);
  float*     logit  = (float*)carve((size_t)Eaug * NHEADS * 4);
  unsigned*  mx     = (unsigned*)carve((size_t)N * NHEADS * 4);
  float*     den    = (float*)carve((size_t)N * NHEADS * 4);
  float*     xlm    = (float*)carve((size_t)N * 4);
  float*     xrm    = (float*)carve((size_t)N * 4);
  float*     xls    = (float*)carve((size_t)N * 4);
  float*     xrs    = (float*)carve((size_t)N * 4);
  float*     lm     = (float*)carve((size_t)Eaug * 4);
  float*     ls     = (float*)carve((size_t)Eaug * 4);
  unsigned*  mxm    = (unsigned*)carve((size_t)N * 4);
  unsigned*  mxs    = (unsigned*)carve((size_t)N * 4);
  float*     denm   = (float*)carve((size_t)N * 4);
  float*     dens   = (float*)carve((size_t)N * 4);
  float*     outm   = (float*)carve((size_t)N * 4);
  float*     outs   = (float*)carve((size_t)N * 4);
  float*     bsum   = (float*)carve((size_t)NBATCH * 4);
  float*     bcnt   = (float*)carve((size_t)NBATCH * 4);

  // ---- self-loop edge_attr = mean of incoming edge_attr ----
  hipMemsetAsync(deg, 0, (size_t)N * 4, stream);
  hipMemsetAsync(ealoop, 0, (size_t)N * 8, stream);
  nmx_deg<<<cdiv(E, 256), 256, 0, stream>>>(dst, eattr, deg, ealoop, E);
  nmx_ealoop_div<<<cdiv(N, 256), 256, 0, stream>>>(deg, ealoop, N);

  // helper: 16-head edge softmax + aggregation into agg
  auto edges16 = [&](const float* We, const float* att) {
    hipMemsetAsync(mx, 0, (size_t)N * NHEADS * 4, stream);
    hipMemsetAsync(den, 0, (size_t)N * NHEADS * 4, stream);
    hipMemsetAsync(agg, 0, (size_t)N * DHID * 4, stream);
    nmx_edge_logit16<<<cdiv((long long)Eaug * 32, 256), 256, 0, stream>>>(
        src, dst, E, Eaug, eattr, ealoop, xl, xr, We, att, logit, mx);
    nmx_edge_soft16<<<cdiv((long long)Eaug * NHEADS, 256), 256, 0, stream>>>(
        dst, E, Eaug, logit, mx, den);
    nmx_edge_agg16<<<cdiv((long long)Eaug * 32, 256), 256, 0, stream>>>(
        src, dst, E, Eaug, logit, den, xl, agg);
  };

  // ---- init GAT layer (D_IN=2 -> 128) + BN + ELU ; writes h and h0 ----
  nmx_init_xlxr<<<cdiv((long long)N * DHID, 256), 256, 0, stream>>>(
      x, iWl, ibl, iWr, ibr, xl, xr, N * DHID);
  edges16(iWe, iatt);
  nmx_bn_elu<<<cdiv((long long)N * DHID, 256), 256, 0, stream>>>(
      agg, ibias, ig, ibe, imn, ivr, nullptr, 0, h, h0, N * DHID);

  // ---- 10 hidden layers: WMMA GEMMs + edge phase + BN/ELU/residual ----
  const int gemmBlocks = cdiv((long long)cdiv(N, 16) * 32, 256);
  for (int l = 0; l < NITER; ++l) {
    nmx_f32_to_f16<<<cdiv((long long)N * DHID, 256), 256, 0, stream>>>(h, hf16, N * DHID);
    nmx_wtrans<<<cdiv(DHID * DHID, 256), 256, 0, stream>>>(hWl + (size_t)l * DHID * DHID, WlT);
    nmx_wtrans<<<cdiv(DHID * DHID, 256), 256, 0, stream>>>(hWr + (size_t)l * DHID * DHID, WrT);
    nmx_gemm_wmma<<<gemmBlocks, 256, 0, stream>>>(hf16, WlT, hbl + (size_t)l * DHID, xl, N);
    nmx_gemm_wmma<<<gemmBlocks, 256, 0, stream>>>(hf16, WrT, hbr + (size_t)l * DHID, xr, N);
    edges16(hWe + (size_t)l * 2 * DHID, hatt + (size_t)l * DHID);
    int addres = ((l + 1) % 2 == 0) ? 1 : 0;
    nmx_bn_elu<<<cdiv((long long)N * DHID, 256), 256, 0, stream>>>(
        agg, hbias + (size_t)l * DHID, hg + (size_t)l * DHID, hbe + (size_t)l * DHID,
        hmn + (size_t)l * DHID, hvr + (size_t)l * DHID, h0, addres, h, nullptr, N * DHID);
  }

  // ---- final mus/sat GATv2 (1 head, 1 channel) ----
  nmx_final_gemv<<<cdiv((long long)N * 32, 256), 256, 0, stream>>>(
      h, mWl, mWr, sWl, sWr, mbl, mbr, sbl, sbr, xlm, xrm, xls, xrs, N);
  hipMemsetAsync(mxm, 0, (size_t)N * 4, stream);
  hipMemsetAsync(mxs, 0, (size_t)N * 4, stream);
  hipMemsetAsync(denm, 0, (size_t)N * 4, stream);
  hipMemsetAsync(dens, 0, (size_t)N * 4, stream);
  hipMemsetAsync(outm, 0, (size_t)N * 4, stream);
  hipMemsetAsync(outs, 0, (size_t)N * 4, stream);
  hipMemsetAsync(bsum, 0, (size_t)NBATCH * 4, stream);
  hipMemsetAsync(bcnt, 0, (size_t)NBATCH * 4, stream);
  nmx_final_logit<<<cdiv(Eaug, 256), 256, 0, stream>>>(
      src, dst, E, Eaug, eattr, ealoop, xlm, xrm, mWe, matt, xls, xrs, sWe, satt,
      lm, ls, mxm, mxs);
  nmx_final_soft<<<cdiv(Eaug, 256), 256, 0, stream>>>(dst, E, Eaug, lm, ls, mxm, mxs, denm, dens);
  nmx_final_agg<<<cdiv(Eaug, 256), 256, 0, stream>>>(
      src, dst, E, Eaug, lm, ls, denm, dens, xlm, xls, outm, outs);

  // ---- outputs: x_mus [N] then x_sat [B] ----
  nmx_out_nodes<<<cdiv(N, 256), 256, 0, stream>>>(
      outm, outs, mbias, sbias, mask, batch, (float*)d_out, bsum, bcnt, N);
  nmx_out_batch<<<1, 32, 0, stream>>>(bsum, bcnt, (float*)d_out, N, NBATCH);
}
